// RVAE_6201932775942
// MI455X (gfx1250) — matrix operand order repaired and checked by reading, not compiled, not run on hardware
//
#include <hip/hip_runtime.h>
#include <hip/hip_bf16.h>
#include <cstddef>

typedef __attribute__((ext_vector_type(16))) _Float16 v16h;
typedef __attribute__((ext_vector_type(8)))  float    v8f;
typedef __attribute__((ext_vector_type(4)))  unsigned int u32x4;
typedef __attribute__((ext_vector_type(8)))  int      i32x8;
typedef __attribute__((ext_vector_type(4)))  int      i32x4;
typedef _Float16 h16;

#define B_   16
#define T_   512
#define F_   257
#define DZ_  16
#define HX_  128
#define HZ_  128
#define HD_  257
#define CC_  32
#define NPIX_ (T_*F_)            // 131584
#define PTILES_ (NPIX_/64)       // 2056 (exact)
#define BTF_  (B_*T_*F_)         // 2105344
#define BTDZ_ (B_*T_*DZ_)        // 131072

#if __has_builtin(__builtin_amdgcn_tensor_load_to_lds) && \
    __has_builtin(__builtin_amdgcn_s_wait_tensorcnt)
#define HAS_TDM 1
#else
#define HAS_TDM 0
#endif

// ---------------------------------------------------------------- WMMA helpers
__device__ __forceinline__ v8f wmma16(v16h a, v16h b, v8f c) {
  return __builtin_amdgcn_wmma_f32_16x16x32_f16(false, a, false, b, (short)0, c,
                                                false, false);
}
__device__ __forceinline__ v8f zero8() {
  v8f z;
  #pragma unroll
  for (int i = 0; i < 8; ++i) z[i] = 0.f;
  return z;
}
// Fragment-major layouts: each 16x32(A)/32x16(B) tile is stored as the exact
// per-lane register image: [32 lanes][16 halves], 32B contiguous per lane.
// A-tile slot for element (m, k) of a 16x32 tile (ISA 7.12.2 16-bit A layout).
__device__ __forceinline__ int a_slot(int m, int k) {
  return ((m + (((k >> 3) & 1) << 4)) << 4) + (((k >> 4) & 1) << 3) + (k & 7);
}
__device__ __forceinline__ v16h load_a_tile(const h16* s) {
  return *(const v16h*)(s + (threadIdx.x & 31) * 16);
}
// B tiles: packed buffer of [KT][NTn] tiles, each 32 lanes x 16 halves.
__device__ __forceinline__ v16h load_b_tile(const h16* Bp, int NTn, int tk, int tn) {
  return *(const v16h*)(Bp + (((size_t)tk * NTn + tn) * 32 + (threadIdx.x & 31)) * 16);
}
__device__ __forceinline__ float sigf(float x) { return 1.f / (1.f + expf(-x)); }

#if HAS_TDM
// TDM 2D tile load: copy `rows` rows of `tile_w8` 8-byte elements from global
// (row stride `stride8` elements) into LDS at `lds_dst`. D# per ISA ch8.
__device__ __forceinline__ void tdm_load_2d(void* lds_dst, const void* gsrc,
                                            unsigned tile_w8, unsigned rows,
                                            unsigned stride8) {
  unsigned lds_off = (unsigned)(size_t)lds_dst;
  unsigned long long ga = (unsigned long long)gsrc;
  u32x4 g0;
  g0[0] = 1u;                                   // count=1, user descriptor
  g0[1] = lds_off;                              // lds_addr [63:32]
  g0[2] = (unsigned)ga;                         // global_addr [95:64]
  g0[3] = ((unsigned)(ga >> 32) & 0x01ffffffu)  // global_addr [120:96]
          | (2u << 30);                         // type=2 ("image")
  i32x8 g1;
  g1[0] = (int)(3u << 16);                      // data_size=8B; no multicast
  g1[1] = (int)((stride8 & 0xffffu) << 16);     // tensor_dim0 lo16
  g1[2] = (int)((stride8 >> 16) | ((rows & 0xffffu) << 16));  // dim0 hi | dim1 lo
  g1[3] = (int)((tile_w8 & 0xffffu) << 16);     // dim1 hi(0) | tile_dim0
  g1[4] = (int)(rows & 0xffffu);                // tile_dim1 | tile_dim2(0)
  g1[5] = (int)stride8;                         // tensor_dim0_stride lo32
  g1[6] = 0;
  g1[7] = 0;
  i32x4 gz;
  gz[0] = 0; gz[1] = 0; gz[2] = 0; gz[3] = 0;
#if __clang_major__ >= 23
  i32x8 gz8;
  #pragma unroll
  for (int i = 0; i < 8; ++i) gz8[i] = 0;
  __builtin_amdgcn_tensor_load_to_lds(g0, g1, gz, gz, gz8, 0);
#else
  __builtin_amdgcn_tensor_load_to_lds(g0, g1, gz, gz, 0);
#endif
  __builtin_amdgcn_s_wait_tensorcnt(0);
}
#endif

// ---------------------------------------------------------------- packing
// Pack W[N,K] fp32 into fragment-major B tiles (KxN logical, K pad Kp, N pad NpTot).
__global__ void k_pack_bfrag(const float* __restrict__ W, h16* __restrict__ Bp,
                             int N, int K, int NpTot, int Kp, int coff, int Nw, int koff) {
  int i = blockIdx.x * blockDim.x + threadIdx.x;
  if (i >= Nw * Kp) return;
  int k = i / Nw, nl = i % Nw;
  int n = coff + nl;
  float v = 0.f;
  if (k >= koff && (k - koff) < K && nl < N) v = W[(size_t)nl * K + (k - koff)];
  int NTn = NpTot >> 4;
  int tk = k >> 5, kk = k & 31;
  int tn = n >> 4, nn = n & 15;
  int lane = nn + (((kk >> 3) & 1) << 4);
  int p = (((kk >> 4) & 1) << 3) + (kk & 7);
  Bp[(((size_t)tk * NTn + tn) * 32 + lane) * 16 + p] = (h16)v;
}

__global__ void k_log16(const float* __restrict__ x, h16* __restrict__ o, int n) {
  int i = blockIdx.x * blockDim.x + threadIdx.x;
  if (i < n) o[i] = (h16)logf(x[i] + 1e-8f);
}

// ---------------------------------------------------------------- conv3x3 (implicit GEMM)
// in:[B][Ci][T][F] f16, wp: fragment-packed [KC][2 tiles], out16:[B][Co][T][F]
// mode: 0 tanh, 1 none, 2 +residual(res16, may alias out16), 3 exp -> outf[B][T][F]
__global__ void k_conv3x3(const h16* __restrict__ in, const h16* __restrict__ wp,
                          const float* __restrict__ bias, h16* out16, const h16* res16,
                          float* outf, int Ci, int Co, int KC, int mode) {
  __shared__ alignas(32) h16 sA[4 * 512];
  int tid = threadIdx.x, w = tid >> 5, lane = tid & 31;
  int b = blockIdx.x / PTILES_, pt = blockIdx.x % PTILES_;
  int p0 = pt * 64 + w * 16;
  int Ktot = Ci * 9;
  h16* myA = sA + w * 512;
  v8f acc0 = zero8(), acc1 = zero8();
  for (int kc = 0; kc < KC; ++kc) {
    if (kc + 1 < KC)
      __builtin_prefetch(wp + ((size_t)(kc + 1) * 2) * 512, 0, 1);
    for (int r = lane; r < 512; r += 32) {
      int m = r >> 5, k = r & 31;
      int kg = kc * 32 + k;
      h16 v = (h16)0.f;
      if (kg < Ktot) {
        int ci = kg / 9, tap = kg % 9, dt = tap / 3 - 1, df = tap % 3 - 1;
        int pix = p0 + m, t = pix / F_, f = pix % F_;
        int tt = t + dt, ff = f + df;
        if (tt >= 0 && tt < T_ && ff >= 0 && ff < F_)
          v = in[(((size_t)b * Ci + ci) * T_ + tt) * F_ + ff];
      }
      myA[a_slot(m, k)] = v;
    }
    v16h a = load_a_tile(myA);
    acc0 = wmma16(a, load_b_tile(wp, 2, kc, 0), acc0);
    if (Co > 16) acc1 = wmma16(a, load_b_tile(wp, 2, kc, 1), acc1);
  }
  int n0 = lane & 15, hf = lane >> 4;
  for (int nt = 0; nt < 2; ++nt) {
    if (nt == 1 && Co <= 16) break;
    int co = nt * 16 + n0;
    if (co >= Co) continue;
    v8f acc = nt ? acc1 : acc0;
    #pragma unroll
    for (int i = 0; i < 8; ++i) {
      int pix = p0 + i + 8 * hf, t = pix / F_, f = pix % F_;
      float v = acc[i] + bias[co];
      size_t oidx = (((size_t)b * Co + co) * T_ + t) * F_ + f;
      if (mode == 0) v = tanhf(v);
      else if (mode == 2) v += (float)res16[oidx];
      if (mode == 3) outf[((size_t)b * T_ + t) * F_ + f] = expf(v);
      else out16[oidx] = (h16)v;
    }
  }
}

// ---------------------------------------------------------------- big GEMM: gi = X @ W^T + b
// X:[B][T][K] f16, row m = t*16+b, time index = rev ? T-1-t : t
// out:[M=8192][N] f32. B panel (KC rows x 4 tiles) staged into LDS by the TDM
// once per block, then shared by all 4 waves (ds_load_b128 reads).
__global__ void k_gemm_xw(const h16* __restrict__ X, const h16* __restrict__ Bp,
                          const float* __restrict__ bias, float* __restrict__ out,
                          int K, int KC, int N, int NTn, int rev) {
  __shared__ alignas(32) h16 sA[4 * 512];
  __shared__ alignas(32) h16 swB[9 * 4 * 512];   // up to 36KB B panel
  int tid = threadIdx.x, w = tid >> 5, lane = tid & 31;
  int mblk = blockIdx.x, nb = blockIdx.y;
#if HAS_TDM
  if (w == 0)
    tdm_load_2d(swB, Bp + ((size_t)nb * 4) * 512, /*tile_w8=*/512, /*rows=*/KC,
                /*stride8=*/(unsigned)NTn * 128);
#else
  for (int i = tid; i < KC * 2048; i += 128) {
    int kc = i >> 11, j = i & 2047;
    swB[i] = Bp[((size_t)kc * NTn + nb * 4) * 512 + j];
  }
#endif
  __syncthreads();
  h16* myA = sA + w * 512;
  v8f acc[4];
  #pragma unroll
  for (int c = 0; c < 4; ++c) acc[c] = zero8();
  for (int kc = 0; kc < KC; ++kc) {
    for (int r = lane; r < 512; r += 32) {
      int m = r >> 5, k = r & 31;
      int mg = mblk * 64 + w * 16 + m, t = mg >> 4, b = mg & 15;
      int tt = rev ? (T_ - 1 - t) : t, kg = kc * 32 + k;
      myA[a_slot(m, k)] = (kg < K) ? X[((size_t)b * T_ + tt) * K + kg] : (h16)0.f;
    }
    v16h a = load_a_tile(myA);
    #pragma unroll
    for (int nt = 0; nt < 4; ++nt)
      acc[nt] = wmma16(a, load_a_tile(swB + (kc * 4 + nt) * 512), acc[nt]);
  }
  int n0 = lane & 15, hf = lane >> 4;
  for (int nt = 0; nt < 4; ++nt) {
    int n = nb * 64 + nt * 16 + n0;
    if (n >= N) continue;
    #pragma unroll
    for (int i = 0; i < 8; ++i)
      out[(size_t)(mblk * 64 + w * 16 + i + 8 * hf) * N + n] = acc[nt][i] + bias[n];
  }
}

// ---------------------------------------------------------------- sequential GRU scan
// One workgroup (8 waves). Per step: gh = h @ Whh^T via WMMA, then gate math.
// gi:[T][16][3H] f32, whh fragment-packed [KC][NT], out f16 at to*os_t + b*os_b + off + j
__global__ void k_scan_gru(const float* __restrict__ gi, const h16* __restrict__ whh,
                           const float* __restrict__ bhh, h16* __restrict__ out,
                           int H, int KC, int NT,
                           int os_t, int os_b, int off, int rev_out) {
  __shared__ alignas(32) h16 shf[9 * 512];   // h state, fragment layout per K chunk
  __shared__ h16 sgh[16 * 832];              // gh staging
  int tid = threadIdx.x, w = tid >> 5, lane = tid & 31;
  int n0 = lane & 15, hf = lane >> 4;
  for (int i = tid; i < 9 * 512; i += 256) shf[i] = (h16)0.f;
  __syncthreads();
  int N3 = 3 * H;
  for (int t = 0; t < T_; ++t) {
    v8f acc[7];
    int cnt = 0;
    for (int nt = w; nt < NT; nt += 8) acc[cnt++] = zero8();
    for (int kc = 0; kc < KC; ++kc) {
      v16h a = load_a_tile(shf + kc * 512);
      int c = 0;
      for (int nt = w; nt < NT; nt += 8) {
        acc[c] = wmma16(a, load_b_tile(whh, NT, kc, nt), acc[c]);
        ++c;
      }
    }
    int c = 0;
    for (int nt = w; nt < NT; nt += 8) {
      #pragma unroll
      for (int i = 0; i < 8; ++i)
        sgh[(i + 8 * hf) * 832 + nt * 16 + n0] = (h16)acc[c][i];
      ++c;
    }
    __syncthreads();
    const float* gr = gi + ((size_t)t * 16) * N3;
    for (int idx = tid; idx < 16 * H; idx += 256) {
      int b = idx / H, j = idx % H;
      float ghr = (float)sgh[b * 832 + j]         + bhh[j];
      float ghz = (float)sgh[b * 832 + H + j]     + bhh[H + j];
      float ghn = (float)sgh[b * 832 + 2 * H + j] + bhh[2 * H + j];
      const float* g = gr + (size_t)b * N3;
      float r = sigf(g[j] + ghr);
      float u = sigf(g[H + j] + ghz);
      float nn = tanhf(g[2 * H + j] + r * ghn);
      int slot = (j >> 5) * 512 + a_slot(b, j & 31);
      float hnew = (1.f - u) * nn + u * (float)shf[slot];
      shf[slot] = (h16)hnew;
      int to = rev_out ? (T_ - 1 - t) : t;
      out[(size_t)to * os_t + (size_t)b * os_b + off + j] = (h16)hnew;
    }
    __syncthreads();
  }
}

// ---------------------------------------------------------------- latent scan (z-GRU + MLPs + reparam)
__global__ void k_scan_latent(const h16* __restrict__ ht16,
                              const h16* __restrict__ wihz, const h16* __restrict__ whhz,
                              const float* __restrict__ bihz, const float* __restrict__ bhhz,
                              const h16* __restrict__ w1cat,
                              const float* __restrict__ b1m, const float* __restrict__ b1v,
                              const h16* __restrict__ w2cat,
                              const float* __restrict__ b2m, const float* __restrict__ b2v,
                              const float* __restrict__ eps,
                              float* __restrict__ zm_o, float* __restrict__ zlv_o,
                              float* __restrict__ z_o, h16* __restrict__ z16) {
  __shared__ alignas(32) h16 szf[512];        // z state (1 chunk, K pad 32)
  __shared__ alignas(32) h16 shf[4 * 512];    // z-GRU h state (K=128)
  __shared__ alignas(32) h16 sAq[12 * 512];   // ht row [hx | h] fragment layout (K=384)
  __shared__ alignas(32) h16 sm1f[8 * 512];   // MLP hidden (K=256)
  __shared__ h16 sgi[16 * 384];
  __shared__ h16 sgh[16 * 384];
  __shared__ float szm[256];
  __shared__ float szv[256];
  int tid = threadIdx.x, w = tid >> 5, lane = tid & 31;
  int n0 = lane & 15, hf = lane >> 4;
  for (int i = tid; i < 512; i += 256) szf[i] = (h16)0.f;
  for (int i = tid; i < 4 * 512; i += 256) shf[i] = (h16)0.f;
  __syncthreads();
  for (int t = 0; t < T_; ++t) {
    // ---- z-GRU: gi (K=16 pad 32) and gh (K=128)
    v8f agi[3], agh[3];
    #pragma unroll
    for (int c = 0; c < 3; ++c) { agi[c] = zero8(); agh[c] = zero8(); }
    {
      v16h az = load_a_tile(szf);
      int c = 0;
      for (int nt = w; nt < 24; nt += 8) {
        agi[c] = wmma16(az, load_b_tile(wihz, 24, 0, nt), agi[c]);
        ++c;
      }
    }
    for (int kc = 0; kc < 4; ++kc) {
      v16h a = load_a_tile(shf + kc * 512);
      int c = 0;
      for (int nt = w; nt < 24; nt += 8) {
        agh[c] = wmma16(a, load_b_tile(whhz, 24, kc, nt), agh[c]);
        ++c;
      }
    }
    {
      int c = 0;
      for (int nt = w; nt < 24; nt += 8) {
        #pragma unroll
        for (int i = 0; i < 8; ++i) {
          int m = i + 8 * hf, n = nt * 16 + n0;
          sgi[m * 384 + n] = (h16)agi[c][i];
          sgh[m * 384 + n] = (h16)agh[c][i];
        }
        ++c;
      }
    }
    // stage hx part of ht row into fragment LDS (cols 0..255)
    for (int idx = tid; idx < 16 * 256; idx += 256) {
      int m = idx >> 8, k = idx & 255;
      sAq[(k >> 5) * 512 + a_slot(m, k & 31)] = ht16[((size_t)t * 16 + m) * 384 + k];
    }
    __syncthreads();
    // ---- gates (h update; write h into both shf and sAq cols 256..383)
    for (int idx = tid; idx < 16 * 128; idx += 256) {
      int b = idx >> 7, j = idx & 127;
      float r = sigf((float)sgi[b * 384 + j] + bihz[j] +
                     (float)sgh[b * 384 + j] + bhhz[j]);
      float u = sigf((float)sgi[b * 384 + 128 + j] + bihz[128 + j] +
                     (float)sgh[b * 384 + 128 + j] + bhhz[128 + j]);
      float nn = tanhf((float)sgi[b * 384 + 256 + j] + bihz[256 + j] +
                       r * ((float)sgh[b * 384 + 256 + j] + bhhz[256 + j]));
      int slot = (j >> 5) * 512 + a_slot(b, j & 31);
      float hnew = (1.f - u) * nn + u * (float)shf[slot];
      shf[slot] = (h16)hnew;
      sAq[(8 + (j >> 5)) * 512 + a_slot(b, j & 31)] = (h16)hnew;
    }
    __syncthreads();
    // ---- MLP layer 1 (both heads, N=256, K=384)
    v8f am[2];
    am[0] = zero8(); am[1] = zero8();
    for (int kc = 0; kc < 12; ++kc) {
      v16h a = load_a_tile(sAq + kc * 512);
      int c = 0;
      for (int nt = w; nt < 16; nt += 8) {
        am[c] = wmma16(a, load_b_tile(w1cat, 16, kc, nt), am[c]);
        ++c;
      }
    }
    {
      int c = 0;
      for (int nt = w; nt < 16; nt += 8) {
        #pragma unroll
        for (int i = 0; i < 8; ++i) {
          int m = i + 8 * hf, n = nt * 16 + n0;
          float bb = (n < 128) ? b1m[n] : b1v[n - 128];
          sm1f[(n >> 5) * 512 + a_slot(m, n & 31)] = (h16)tanhf(am[c][i] + bb);
        }
        ++c;
      }
    }
    __syncthreads();
    // ---- MLP layer 2 (N=32: [zmean|zlogvar], block-diag K=256)
    if (w < 2) {
      v8f a2 = zero8();
      for (int kc = 0; kc < 8; ++kc)
        a2 = wmma16(load_a_tile(sm1f + kc * 512), load_b_tile(w2cat, 2, kc, w), a2);
      #pragma unroll
      for (int i = 0; i < 8; ++i) {
        int m = i + 8 * hf;
        if (w == 0) szm[m * 16 + n0] = a2[i] + b2m[n0];
        else        szv[m * 16 + n0] = a2[i] + b2v[n0];
      }
    }
    __syncthreads();
    // ---- reparametrization + outputs
    {
      int b = tid >> 4, j = tid & 15;
      float zm = szm[b * 16 + j], zlv = szv[b * 16 + j];
      float e = eps[((size_t)t * 16 + b) * 16 + j];
      float zn = zm + e * expf(0.5f * zlv);
      size_t o = ((size_t)b * T_ + t) * 16 + j;
      zm_o[o] = zm; zlv_o[o] = zlv; z_o[o] = zn;
      z16[o] = (h16)zn;
      szf[a_slot(b, j)] = (h16)zn;
    }
    __syncthreads();
  }
}

// ---------------------------------------------------------------- host
enum {
  I_X = 0, I_PREE_W, I_PREE_B,
  I_RE0_W1, I_RE0_B1, I_RE0_W2, I_RE0_B2,
  I_RE1_W1, I_RE1_B1, I_RE1_W2, I_RE1_B2,
  I_POSTE_W, I_POSTE_B,
  I_GEF_WIH, I_GEF_WHH, I_GEF_BIH, I_GEF_BHH,
  I_GEB_WIH, I_GEB_WHH, I_GEB_BIH, I_GEB_BHH,
  I_GZ_WIH, I_GZ_WHH, I_GZ_BIH, I_GZ_BHH,
  I_M1M_W, I_M1M_B, I_M2M_W, I_M2M_B,
  I_M1V_W, I_M1V_B, I_M2V_W, I_M2V_B,
  I_GDF_WIH, I_GDF_WHH, I_GDF_BIH, I_GDF_BHH,
  I_GDB_WIH, I_GDB_WHH, I_GDB_BIH, I_GDB_BHH,
  I_PRED_W, I_PRED_B,
  I_RD0_W1, I_RD0_B1, I_RD0_W2, I_RD0_B2,
  I_RD1_W1, I_RD1_B1, I_RD1_W2, I_RD1_B2,
  I_POSTD_W, I_POSTD_B,
  I_EPS
};

extern "C" void kernel_launch(void* const* d_in, const int* in_sizes, int n_in,
                              void* d_out, int out_size, void* d_ws, size_t ws_size,
                              hipStream_t stream) {
  (void)in_sizes; (void)n_in; (void)out_size; (void)ws_size;
  auto F32 = [&](int i) { return (const float*)d_in[i]; };

  size_t cur = 0;
  auto alloc = [&](size_t bytes) -> char* {
    cur = (cur + 255) & ~(size_t)255;
    char* p = (char*)d_ws + cur;
    cur += bytes;
    return p;
  };
  h16* bufA = (h16*)alloc((size_t)B_ * CC_ * T_ * F_ * 2);
  h16* bufB = (h16*)alloc((size_t)B_ * CC_ * T_ * F_ * 2);
  h16* xt16 = (h16*)alloc((size_t)BTF_ * 2);
  h16* ht16 = (h16*)alloc((size_t)T_ * 16 * 384 * 2);
  h16* z16  = (h16*)alloc((size_t)BTDZ_ * 2);
  h16* hd16 = (h16*)alloc((size_t)B_ * 2 * T_ * HD_ * 2);
  float* giPool = (float*)alloc((size_t)2 * T_ * 16 * 771 * 4);
  float* giEF = giPool;
  float* giEB = giPool + (size_t)T_ * 16 * 384;
  float* giDF = giPool;
  float* giDB = giPool + (size_t)T_ * 16 * 771;
  h16* wpPreE  = (h16*)alloc(32 * 32 * 2);
  h16* wpRE[4]; for (int i = 0; i < 4; ++i) wpRE[i] = (h16*)alloc(288 * 32 * 2);
  h16* wpPostE = (h16*)alloc(288 * 32 * 2);
  h16* wpPreD  = (h16*)alloc(32 * 32 * 2);
  h16* wpRD[4]; for (int i = 0; i < 4; ++i) wpRD[i] = (h16*)alloc(288 * 32 * 2);
  h16* wpPostD = (h16*)alloc(288 * 32 * 2);
  h16* pWihEF = (h16*)alloc(288 * 384 * 2);
  h16* pWihEB = (h16*)alloc(288 * 384 * 2);
  h16* pWhhEF = (h16*)alloc(128 * 384 * 2);
  h16* pWhhEB = (h16*)alloc(128 * 384 * 2);
  h16* pWihZ  = (h16*)alloc(32 * 384 * 2);
  h16* pWhhZ  = (h16*)alloc(128 * 384 * 2);
  h16* pW1    = (h16*)alloc(384 * 256 * 2);
  h16* pW2    = (h16*)alloc(256 * 32 * 2);
  h16* pWihDF = (h16*)alloc(32 * 832 * 2);
  h16* pWihDB = (h16*)alloc(32 * 832 * 2);
  h16* pWhhDF = (h16*)alloc(288 * 832 * 2);
  h16* pWhhDB = (h16*)alloc(288 * 832 * 2);

  auto pack = [&](const float* W, h16* Bp, int N, int K, int NpTot, int Kp,
                  int coff, int Nw, int koff) {
    int tot = Nw * Kp;
    k_pack_bfrag<<<dim3((tot + 255) / 256), dim3(256), 0, stream>>>(W, Bp, N, K, NpTot,
                                                                    Kp, coff, Nw, koff);
  };
  // conv weight packs (OIHW flattens to [Co][Ci*9])
  pack(F32(I_PREE_W), wpPreE, 32, 9, 32, 32, 0, 32, 0);
  pack(F32(I_RE0_W1), wpRE[0], 32, 288, 32, 288, 0, 32, 0);
  pack(F32(I_RE0_W2), wpRE[1], 32, 288, 32, 288, 0, 32, 0);
  pack(F32(I_RE1_W1), wpRE[2], 32, 288, 32, 288, 0, 32, 0);
  pack(F32(I_RE1_W2), wpRE[3], 32, 288, 32, 288, 0, 32, 0);
  pack(F32(I_POSTE_W), wpPostE, 1, 288, 32, 288, 0, 32, 0);
  pack(F32(I_PRED_W), wpPreD, 32, 18, 32, 32, 0, 32, 0);
  pack(F32(I_RD0_W1), wpRD[0], 32, 288, 32, 288, 0, 32, 0);
  pack(F32(I_RD0_W2), wpRD[1], 32, 288, 32, 288, 0, 32, 0);
  pack(F32(I_RD1_W1), wpRD[2], 32, 288, 32, 288, 0, 32, 0);
  pack(F32(I_RD1_W2), wpRD[3], 32, 288, 32, 288, 0, 32, 0);
  pack(F32(I_POSTD_W), wpPostD, 1, 288, 32, 288, 0, 32, 0);
  // GRU / MLP packs
  pack(F32(I_GEF_WIH), pWihEF, 384, 257, 384, 288, 0, 384, 0);
  pack(F32(I_GEB_WIH), pWihEB, 384, 257, 384, 288, 0, 384, 0);
  pack(F32(I_GEF_WHH), pWhhEF, 384, 128, 384, 128, 0, 384, 0);
  pack(F32(I_GEB_WHH), pWhhEB, 384, 128, 384, 128, 0, 384, 0);
  pack(F32(I_GZ_WIH), pWihZ, 384, 16, 384, 32, 0, 384, 0);
  pack(F32(I_GZ_WHH), pWhhZ, 384, 128, 384, 128, 0, 384, 0);
  pack(F32(I_M1M_W), pW1, 128, 384, 256, 384, 0, 128, 0);
  pack(F32(I_M1V_W), pW1, 128, 384, 256, 384, 128, 128, 0);
  pack(F32(I_M2M_W), pW2, 16, 128, 32, 256, 0, 16, 0);
  pack(F32(I_M2V_W), pW2, 16, 128, 32, 256, 16, 16, 128);
  pack(F32(I_GDF_WIH), pWihDF, 771, 16, 832, 32, 0, 832, 0);
  pack(F32(I_GDB_WIH), pWihDB, 771, 16, 832, 32, 0, 832, 0);
  pack(F32(I_GDF_WHH), pWhhDF, 771, 257, 832, 288, 0, 832, 0);
  pack(F32(I_GDB_WHH), pWhhDB, 771, 257, 832, 288, 0, 832, 0);

  auto conv = [&](const h16* in, const h16* wp, const float* bias, h16* o16,
                  const h16* res, float* of, int Ci, int Co, int KC, int mode) {
    k_conv3x3<<<dim3(B_ * PTILES_), dim3(128), 0, stream>>>(in, wp, bias, o16, res, of,
                                                            Ci, Co, KC, mode);
  };
  auto gemm = [&](const h16* X, const h16* Bp, const float* bias, float* o,
                  int K, int KC, int N, int NTn, int rev) {
    k_gemm_xw<<<dim3(128, (N + 63) / 64), dim3(128), 0, stream>>>(X, Bp, bias, o,
                                                                  K, KC, N, NTn, rev);
  };

  float* out   = (float*)d_out;
  float* zm_o  = out + BTF_;
  float* zlv_o = zm_o + BTDZ_;
  float* z_o   = zlv_o + BTDZ_;

  // ---------------- encoder conv stack ----------------
  k_log16<<<dim3((BTF_ + 255) / 256), dim3(256), 0, stream>>>(F32(I_X), bufB, BTF_);
  conv(bufB, wpPreE, F32(I_PREE_B), bufA, nullptr, nullptr, 1, 32, 1, 0);
  conv(bufA, wpRE[0], F32(I_RE0_B1), bufB, nullptr, nullptr, 32, 32, 9, 0);
  conv(bufB, wpRE[1], F32(I_RE0_B2), bufA, bufA, nullptr, 32, 32, 9, 2);
  conv(bufA, wpRE[2], F32(I_RE1_B1), bufB, nullptr, nullptr, 32, 32, 9, 0);
  conv(bufB, wpRE[3], F32(I_RE1_B2), bufA, bufA, nullptr, 32, 32, 9, 2);
  conv(bufA, wpPostE, F32(I_POSTE_B), xt16, nullptr, nullptr, 32, 1, 9, 1);
  // ---------------- encoder bi-GRU (over time-reversed input) ----------------
  gemm(xt16, pWihEF, F32(I_GEF_BIH), giEF, 257, 9, 384, 24, 1);
  gemm(xt16, pWihEB, F32(I_GEB_BIH), giEB, 257, 9, 384, 24, 0);
  k_scan_gru<<<dim3(1), dim3(256), 0, stream>>>(giEF, pWhhEF, F32(I_GEF_BHH), ht16,
                                                128, 4, 24, 16 * 384, 384, 0, 1);
  k_scan_gru<<<dim3(1), dim3(256), 0, stream>>>(giEB, pWhhEB, F32(I_GEB_BHH), ht16,
                                                128, 4, 24, 16 * 384, 384, 128, 0);
  // ---------------- latent scan ----------------
  k_scan_latent<<<dim3(1), dim3(256), 0, stream>>>(
      ht16, pWihZ, pWhhZ, F32(I_GZ_BIH), F32(I_GZ_BHH),
      pW1, F32(I_M1M_B), F32(I_M1V_B), pW2, F32(I_M2M_B), F32(I_M2V_B),
      F32(I_EPS), zm_o, zlv_o, z_o, z16);
  // ---------------- decoder bi-GRU ----------------
  gemm(z16, pWihDF, F32(I_GDF_BIH), giDF, 16, 1, 771, 52, 0);
  gemm(z16, pWihDB, F32(I_GDB_BIH), giDB, 16, 1, 771, 52, 1);
  k_scan_gru<<<dim3(1), dim3(256), 0, stream>>>(giDF, pWhhDF, F32(I_GDF_BHH), hd16,
                                                257, 9, 52, HD_, 2 * T_ * HD_, 0, 0);
  k_scan_gru<<<dim3(1), dim3(256), 0, stream>>>(giDB, pWhhDB, F32(I_GDB_BHH), hd16,
                                                257, 9, 52, HD_, 2 * T_ * HD_,
                                                T_ * HD_, 1);
  // ---------------- decoder conv stack ----------------
  conv(hd16, wpPreD, F32(I_PRED_B), bufA, nullptr, nullptr, 2, 32, 1, 0);
  conv(bufA, wpRD[0], F32(I_RD0_B1), bufB, nullptr, nullptr, 32, 32, 9, 0);
  conv(bufB, wpRD[1], F32(I_RD0_B2), bufA, bufA, nullptr, 32, 32, 9, 2);
  conv(bufA, wpRD[2], F32(I_RD1_B1), bufB, nullptr, nullptr, 32, 32, 9, 0);
  conv(bufB, wpRD[3], F32(I_RD1_B2), bufA, bufA, nullptr, 32, 32, 9, 2);
  conv(bufA, wpPostD, F32(I_POSTD_B), nullptr, nullptr, out, 32, 1, 9, 3);
}
